// graphNetwork_46145128628936
// MI455X (gfx1250) — matrix operand order repaired and checked by reading, not compiled, not run on hardware
//
#include <hip/hip_runtime.h>

typedef __attribute__((ext_vector_type(16))) _Float16 v16h;
typedef __attribute__((ext_vector_type(8)))  float    v8f;

#define EPS_LN 1e-5f
#define HSTEP  0.1f

__device__ __forceinline__ v8f wmma_f16(v16h a, v16h b, v8f c) {
  // emits v_wmma_f32_16x16x32_f16
  return __builtin_amdgcn_wmma_f32_16x16x32_f16(false, a, false, b, (short)0, c, false, false);
}

// ---------------------------------------------------------------------------
// Pack a row-major fp32 weight (O x K, O%16==0, K%32==0) into WMMA A-fragment
// order (f16).  Fragment element (mt, kc, lane, j) stored linearly; K mapping
// contiguous: k = 16*(lane/16) + j (matches B operand reads).
// ---------------------------------------------------------------------------
__global__ void k_pack_frag(const float* __restrict__ W, _Float16* __restrict__ F,
                            int O, int K) {
  int t = blockIdx.x * 256 + threadIdx.x;
  int total = O * K;
  if (t >= total) return;
  int j    = t & 15;
  int lane = (t >> 4) & 31;
  int kcN  = K >> 5;
  int kc   = (t >> 9) % kcN;
  int mt   = t / (512 * kcN);
  int m  = mt * 16 + (lane & 15);
  int kk = kc * 32 + ((lane >= 16) ? 16 : 0) + j;
  F[t] = (_Float16)W[m * K + kk];
}

// ---------------------------------------------------------------------------
// Open projection: Y(64 x M) = W(64 x 16) * X(16 x M).  K=16 -> memory bound,
// plain VALU kernel with weights staged in LDS.
// ---------------------------------------------------------------------------
__global__ void k_open(const float* __restrict__ W, const float* __restrict__ X,
                       float* __restrict__ Y, long long M) {
  __shared__ float sW[64 * 16];
  for (int t = threadIdx.x; t < 1024; t += 256) sW[t] = W[t];
  __syncthreads();
  long long m = (long long)blockIdx.x * 256 + threadIdx.x;
  if (m >= M) return;
  float x[16];
#pragma unroll
  for (int i = 0; i < 16; ++i) x[i] = X[(long long)i * M + m];
#pragma unroll
  for (int o = 0; o < 64; ++o) {
    float s = 0.f;
#pragma unroll
    for (int i = 0; i < 16; ++i) s = fmaf(sW[o * 16 + i], x[i], s);
    Y[(long long)o * M + m] = s;
  }
}

__global__ void k_zero(float* p, long long n) {
  long long i = (long long)blockIdx.x * 256 + threadIdx.x;
  long long stride = (long long)gridDim.x * 256;
  for (; i < n; i += stride) p[i] = 0.f;
}

__global__ void k_finalize(float* stats, int sumSlot, int outSlot, float count) {
  float s = stats[sumSlot], sq = stats[sumSlot + 1];
  float mean = s / count;
  float var  = sq / count - mean * mean;
  stats[outSlot]     = mean;
  stats[outSlot + 1] = rsqrtf(var + EPS_LN);
  stats[sumSlot] = 0.f;
  stats[sumSlot + 1] = 0.f;
}

// ---------------------------------------------------------------------------
// Fused gather + concat([intX, xe, gradX]) + GEMM KE1(64x192) + LN statistics.
// One wave per 16-edge tile; 8 waves / block.  Output y1 stored f16 (pure
// scratch, consumed by the next WMMA stage anyway) to halve HBM traffic.
// ---------------------------------------------------------------------------
__global__ __launch_bounds__(256) void k_edge1(
    const float* __restrict__ xn, const float* __restrict__ xe,
    const int* __restrict__ iInd, const int* __restrict__ jInd,
    const _Float16* __restrict__ Afrag, _Float16* __restrict__ y1,
    float* __restrict__ stats, long long N, long long E) {
  __shared__ __attribute__((aligned(32))) _Float16 lds[8 * 16 * 192];
  __shared__ float ssum[2];
  if (threadIdx.x < 2) ssum[threadIdx.x] = 0.f;

  int lane = threadIdx.x & 31, wave = threadIdx.x >> 5;
  _Float16* tile = lds + wave * (16 * 192);
  long long nt   = (long long)blockIdx.x * 8 + wave;
  long long maxt = (E >> 4) - 1;
  bool valid = nt <= maxt;
  if (!valid) nt = maxt;
  int e16 = lane & 15, hi = lane >> 4;
  long long e = nt * 16 + e16;
  int ii = iInd[e], jj = jInd[e];

  for (int r = hi; r < 64; r += 2) {
    float xi = xn[(long long)r * N + ii];
    float xj = xn[(long long)r * N + jj];
    float xv = xe[(long long)r * E + e];
    __builtin_prefetch(&xe[(long long)r * E + e + 2048], 0, 1);
    tile[e16 * 192 + r]       = (_Float16)(0.5f * (xi + xj));
    tile[e16 * 192 + 64 + r]  = (_Float16)xv;
    tile[e16 * 192 + 128 + r] = (_Float16)(xi - xj);
  }
  __syncthreads();

  v8f acc[4] = {};
#pragma unroll
  for (int kc = 0; kc < 6; ++kc) {
    v16h b = *(const v16h*)(tile + e16 * 192 + kc * 32 + hi * 16);
#pragma unroll
    for (int mt = 0; mt < 4; ++mt) {
      v16h a = *(const v16h*)(Afrag + ((mt * 6 + kc) * 32 + lane) * 16);
      acc[mt] = wmma_f16(a, b, acc[mt]);
    }
  }

  float s = 0.f, sq = 0.f;
  if (valid) {
#pragma unroll
    for (int mt = 0; mt < 4; ++mt)
#pragma unroll
      for (int r = 0; r < 8; ++r) {
        float v = acc[mt][r];
        int m = mt * 16 + r + hi * 8;
        y1[(long long)m * E + e] = (_Float16)v;
        s += v; sq += v * v;
      }
  }
  atomicAdd(&ssum[0], s);
  atomicAdd(&ssum[1], sq);
  __syncthreads();
  if (threadIdx.x == 0) {
    atomicAdd(&stats[0], ssum[0]);
    atomicAdd(&stats[1], ssum[1]);
  }
}

// ---------------------------------------------------------------------------
// Generic 64x64 GEMM over an M-column stream (f32 or f16 input) with optional
// fused (global-LN + ReLU) on load, optional LN-stat accumulation, and output
// mode: 0 = store f32, 1 = residual add (scaled) f32, 2 = store f16.
// Used for KE2 (f16->f16 + stats), KE3 (f16->residual f32), KEclose, KNclose.
// ---------------------------------------------------------------------------
template <typename TIN>
__global__ __launch_bounds__(256) void k_gemm64(
    const TIN* __restrict__ in, const _Float16* __restrict__ Afrag,
    float* __restrict__ outF, _Float16* __restrict__ outH,
    const float* __restrict__ stats, int lnSlot,
    float* __restrict__ statsOut, float outScale, int outMode, long long M) {
  __shared__ __attribute__((aligned(32))) _Float16 lds[8 * 16 * 64];
  __shared__ float ssum[2];
  if (threadIdx.x < 2) ssum[threadIdx.x] = 0.f;

  int lane = threadIdx.x & 31, wave = threadIdx.x >> 5;
  _Float16* tile = lds + wave * (16 * 64);
  long long nt   = (long long)blockIdx.x * 8 + wave;
  long long maxt = (M >> 4) - 1;
  bool valid = nt <= maxt;
  if (!valid) nt = maxt;
  int e16 = lane & 15, hi = lane >> 4;
  long long c = nt * 16 + e16;

  float mean = 0.f, inv = 1.f;
  if (lnSlot >= 0) { mean = stats[lnSlot]; inv = stats[lnSlot + 1]; }

  for (int r = hi; r < 64; r += 2) {
    float v = (float)in[(long long)r * M + c];
    __builtin_prefetch(&in[(long long)r * M + c + 2048], 0, 1);
    if (lnSlot >= 0) v = fmaxf(0.f, (v - mean) * inv);
    tile[e16 * 64 + r] = (_Float16)v;
  }
  __syncthreads();

  v8f acc[4] = {};
#pragma unroll
  for (int kc = 0; kc < 2; ++kc) {
    v16h b = *(const v16h*)(tile + e16 * 64 + kc * 32 + hi * 16);
#pragma unroll
    for (int mt = 0; mt < 4; ++mt) {
      v16h a = *(const v16h*)(Afrag + ((mt * 2 + kc) * 32 + lane) * 16);
      acc[mt] = wmma_f16(a, b, acc[mt]);
    }
  }

  float s = 0.f, sq = 0.f;
  if (valid) {
#pragma unroll
    for (int mt = 0; mt < 4; ++mt)
#pragma unroll
      for (int r = 0; r < 8; ++r) {
        float v = acc[mt][r] * outScale;
        int m = mt * 16 + r + hi * 8;
        long long idx = (long long)m * M + c;
        if (outMode == 2)      outH[idx] = (_Float16)v;
        else if (outMode == 1) outF[idx] += v;
        else                   outF[idx] = v;
        s += v; sq += v * v;
      }
  }
  if (statsOut) {
    atomicAdd(&ssum[0], s);
    atomicAdd(&ssum[1], sq);
    __syncthreads();
    if (threadIdx.x == 0) {
      atomicAdd(&statsOut[0], ssum[0]);
      atomicAdd(&statsOut[1], ssum[1]);
    }
  }
}

// ---------------------------------------------------------------------------
// Scatter: divXe += +/- xec, intXe += 0.5*xec at iInd/jInd (xec = LN+ReLU(y2)).
// y2 is f16 scratch.
// ---------------------------------------------------------------------------
__global__ void k_scatter(const _Float16* __restrict__ y2, const float* __restrict__ stats,
                          const int* __restrict__ iInd, const int* __restrict__ jInd,
                          float* __restrict__ intXe, float* __restrict__ divXe,
                          long long N, long long E) {
  long long e = (long long)blockIdx.x * 256 + threadIdx.x;
  if (e >= E) return;
  float mean = stats[6], inv = stats[7];
  int ii = iInd[e], jj = jInd[e];
  for (int c = 0; c < 64; ++c) {
    float v = (float)y2[(long long)c * E + e];
    __builtin_prefetch(&y2[(long long)c * E + e + 4096], 0, 1);
    v = fmaxf(0.f, (v - mean) * inv);
    atomicAdd(&divXe[(long long)c * N + ii],  v);
    atomicAdd(&divXe[(long long)c * N + jj], -v);
    atomicAdd(&intXe[(long long)c * N + ii], 0.5f * v);
    atomicAdd(&intXe[(long long)c * N + jj], 0.5f * v);
  }
}

// ---------------------------------------------------------------------------
// Node update: xn += H * KN(64x192) * concat([intXe, xn, divXe]).
// ---------------------------------------------------------------------------
__global__ __launch_bounds__(256) void k_node(
    const _Float16* __restrict__ Afrag, const float* __restrict__ intXe,
    const float* __restrict__ xn, const float* __restrict__ divXe,
    float* __restrict__ xnOut, long long N) {
  __shared__ __attribute__((aligned(32))) _Float16 lds[8 * 16 * 192];
  int lane = threadIdx.x & 31, wave = threadIdx.x >> 5;
  _Float16* tile = lds + wave * (16 * 192);
  long long nt   = (long long)blockIdx.x * 8 + wave;
  long long maxt = (N >> 4) - 1;
  bool valid = nt <= maxt;
  if (!valid) nt = maxt;
  int e16 = lane & 15, hi = lane >> 4;
  long long c = nt * 16 + e16;

  for (int r = hi; r < 64; r += 2) {
    tile[e16 * 192 + r]       = (_Float16)intXe[(long long)r * N + c];
    tile[e16 * 192 + 64 + r]  = (_Float16)xn[(long long)r * N + c];
    tile[e16 * 192 + 128 + r] = (_Float16)divXe[(long long)r * N + c];
  }
  __syncthreads();

  v8f acc[4] = {};
#pragma unroll
  for (int kc = 0; kc < 6; ++kc) {
    v16h b = *(const v16h*)(tile + e16 * 192 + kc * 32 + hi * 16);
#pragma unroll
    for (int mt = 0; mt < 4; ++mt) {
      v16h a = *(const v16h*)(Afrag + ((mt * 6 + kc) * 32 + lane) * 16);
      acc[mt] = wmma_f16(a, b, acc[mt]);
    }
  }
  if (valid) {
#pragma unroll
    for (int mt = 0; mt < 4; ++mt)
#pragma unroll
      for (int r = 0; r < 8; ++r) {
        int m = mt * 16 + r + hi * 8;
        xnOut[(long long)m * N + c] += HSTEP * acc[mt][r];
      }
  }
}

// ---------------------------------------------------------------------------
extern "C" void kernel_launch(void* const* d_in, const int* in_sizes, int n_in,
                              void* d_out, int out_size, void* d_ws, size_t ws_size,
                              hipStream_t stream) {
  (void)n_in; (void)out_size; (void)ws_size;
  const float* xn0     = (const float*)d_in[0];
  const float* xe0     = (const float*)d_in[1];
  const int*   iInd    = (const int*)d_in[2];
  const int*   jInd    = (const int*)d_in[3];
  const float* KNopen  = (const float*)d_in[4];
  const float* KEopen  = (const float*)d_in[5];
  const float* KNclose = (const float*)d_in[6];
  const float* KEclose = (const float*)d_in[7];
  const float* KE1     = (const float*)d_in[8];
  const float* KE2     = (const float*)d_in[9];
  const float* KE3     = (const float*)d_in[10];
  const float* KN      = (const float*)d_in[11];

  const long long N = in_sizes[0] / 16;
  const long long E = in_sizes[1] / 16;
  const int       L = in_sizes[8] / (64 * 192);

  // ---- workspace bump allocator (256B aligned) ----
  char* wsb = (char*)d_ws;
  size_t off = 0;
  auto alloc = [&](size_t bytes) -> void* {
    void* p = wsb + off;
    off = (off + bytes + 255) & ~(size_t)255;
    return p;
  };
  float*     xn    = (float*)alloc(sizeof(float) * 64 * N);
  float*     xe    = (float*)alloc(sizeof(float) * 64 * E);
  _Float16*  y1    = (_Float16*)alloc(sizeof(_Float16) * 64 * E);
  float*     intXe = (float*)alloc(sizeof(float) * 64 * N);
  float*     divXe = (float*)alloc(sizeof(float) * 64 * N);
  float*     stats = (float*)alloc(sizeof(float) * 8);
  _Float16*  KE1f  = (_Float16*)alloc(sizeof(_Float16) * (size_t)L * 64 * 192);
  _Float16*  KE2f  = (_Float16*)alloc(sizeof(_Float16) * (size_t)L * 64 * 64);
  _Float16*  KE3f  = (_Float16*)alloc(sizeof(_Float16) * (size_t)L * 64 * 64);
  _Float16*  KNf   = (_Float16*)alloc(sizeof(_Float16) * (size_t)L * 64 * 192);
  _Float16*  KNcf  = (_Float16*)alloc(sizeof(_Float16) * 64 * 64);
  _Float16*  KEcf  = (_Float16*)alloc(sizeof(_Float16) * 64 * 64);

  float* outXn = (float*)d_out;
  float* outXe = outXn + 64 * N;
  _Float16* y2 = (_Float16*)outXe;  // reuse xe-output region as f16 scratch

  k_zero<<<1, 256, 0, stream>>>(stats, 8);

  for (int l = 0; l < L; ++l) {
    k_pack_frag<<<(64 * 192 + 255) / 256, 256, 0, stream>>>(KE1 + (size_t)l * 64 * 192,
                                                            KE1f + (size_t)l * 64 * 192, 64, 192);
    k_pack_frag<<<(64 * 64 + 255) / 256, 256, 0, stream>>>(KE2 + (size_t)l * 64 * 64,
                                                           KE2f + (size_t)l * 64 * 64, 64, 64);
    k_pack_frag<<<(64 * 64 + 255) / 256, 256, 0, stream>>>(KE3 + (size_t)l * 64 * 64,
                                                           KE3f + (size_t)l * 64 * 64, 64, 64);
    k_pack_frag<<<(64 * 192 + 255) / 256, 256, 0, stream>>>(KN + (size_t)l * 64 * 192,
                                                            KNf + (size_t)l * 64 * 192, 64, 192);
  }
  k_pack_frag<<<(64 * 64 + 255) / 256, 256, 0, stream>>>(KNclose, KNcf, 64, 64);
  k_pack_frag<<<(64 * 64 + 255) / 256, 256, 0, stream>>>(KEclose, KEcf, 64, 64);

  k_open<<<(int)((N + 255) / 256), 256, 0, stream>>>(KNopen, xn0, xn, N);
  k_open<<<(int)((E + 255) / 256), 256, 0, stream>>>(KEopen, xe0, xe, E);

  const int egrid = (int)((E / 16 + 7) / 8);
  const int ngrid = (int)((N / 16 + 7) / 8);
  const float cnt = (float)(64.0 * (double)E);

  for (int l = 0; l < L; ++l) {
    k_edge1<<<egrid, 256, 0, stream>>>(xn, xe, iInd, jInd,
                                       KE1f + (size_t)l * 64 * 192, y1, stats, N, E);
    k_finalize<<<1, 1, 0, stream>>>(stats, 0, 2, cnt);
    k_gemm64<_Float16><<<egrid, 256, 0, stream>>>(y1, KE2f + (size_t)l * 64 * 64,
                                                  nullptr, y2, stats, 2, stats + 4,
                                                  1.0f, 2, E);
    k_finalize<<<1, 1, 0, stream>>>(stats, 4, 6, cnt);
    k_zero<<<2048, 256, 0, stream>>>(intXe, 64 * N);
    k_zero<<<2048, 256, 0, stream>>>(divXe, 64 * N);
    k_scatter<<<(int)((E + 255) / 256), 256, 0, stream>>>(y2, stats, iInd, jInd,
                                                          intXe, divXe, N, E);
    k_gemm64<_Float16><<<egrid, 256, 0, stream>>>(y2, KE3f + (size_t)l * 64 * 64,
                                                  xe, nullptr, stats, 6, nullptr,
                                                  HSTEP, 1, E);
    k_node<<<ngrid, 256, 0, stream>>>(KNf + (size_t)l * 64 * 192, intXe, xn, divXe, xn, N);
  }

  k_gemm64<float><<<ngrid, 256, 0, stream>>>(xn, KNcf, outXn, nullptr, nullptr, -1,
                                             nullptr, 1.0f, 0, N);
  k_gemm64<float><<<egrid, 256, 0, stream>>>(xe, KEcf, outXe, nullptr, nullptr, -1,
                                             nullptr, 1.0f, 0, E);
}
